// DeepGCN_70858370450166
// MI455X (gfx1250) — matrix operand order repaired and checked by reading, not compiled, not run on hardware
//
#include <hip/hip_runtime.h>

#define NN   20000
#define EE   320000
#define GG   128
#define INC  128
#define HH   64
#define HH2  128
#define LL   8

typedef __attribute__((ext_vector_type(16))) _Float16 v16h;
typedef __attribute__((ext_vector_type(8)))  float    v8f;
typedef __attribute__((ext_vector_type(4)))  int      v4i;
typedef __attribute__((address_space(1))) v4i* gv4i_p;   // global 16B vector ptr
typedef __attribute__((address_space(3))) v4i* lv4i_p;   // LDS 16B vector ptr

#if __has_builtin(__builtin_amdgcn_global_load_async_to_lds_b128) && \
    __has_builtin(__builtin_amdgcn_s_wait_asynccnt)
#define ASYNC_LDS_COPY 1
#else
#define ASYNC_LDS_COPY 0
#endif

// ---------------------------------------------------------------- utilities
__global__ void k_zero(float* __restrict__ p, int n) {
  int i = blockIdx.x * blockDim.x + threadIdx.x;
  if (i < n) p[i] = 0.f;
}

// ------------------------------------------------------- WMMA f16->f32 GEMM
// C[M x Nc] = A[M x K] @ B[K x Nc] (+ bias[Nc]) (+ res).
// M%16==0, K%32==0, Nc % (16*NT)==0, K*Nc*4 bytes of dynamic LDS.
// The block first stages B into LDS with CDNA5 async global->LDS copies
// (GLOBAL_LOAD_ASYNC_TO_LDS_B128 / s_wait_asynccnt), then each wave32
// computes a 16 x (16*NT) strip: A fragment loaded once (4x b128) and
// reused across NT v_wmma_f32_16x16x32_f16 issues per K-step; B fragments
// come from LDS.
template <bool HAS_BIAS, bool HAS_RES, int NT>
__global__ void k_gemm_wmma(const float* __restrict__ A, const float* __restrict__ B,
                            const float* __restrict__ bias, const float* __restrict__ res,
                            float* __restrict__ C, int M, int K, int Nc) {
  extern __shared__ float Blds[];

  // ---- stage B (K*Nc floats) into LDS, 16B per lane per op
  int total4 = (K * Nc) >> 2;
  for (int i = threadIdx.x; i < total4; i += blockDim.x) {
#if ASYNC_LDS_COPY
    __builtin_amdgcn_global_load_async_to_lds_b128(
        (gv4i_p)(B + 4 * (size_t)i),
        (lv4i_p)(Blds + 4 * (size_t)i), 0, 0);
#else
    *(float4*)(Blds + 4 * (size_t)i) = *(const float4*)(B + 4 * (size_t)i);
#endif
  }
#if ASYNC_LDS_COPY
  __builtin_amdgcn_s_wait_asynccnt(0);
#endif
  __syncthreads();

  int wave = (blockIdx.x * blockDim.x + threadIdx.x) >> 5;
  int lane = threadIdx.x & 31;
  int ngroups = Nc / (16 * NT);
  int mt = wave / ngroups;
  int ng = wave - mt * ngroups;
  if (mt * 16 >= M) return;            // wave-uniform: EXEC stays all-ones for WMMA
  int hsel = lane >> 4;                // 0: lanes 0-15, 1: lanes 16-31
  int rm   = lane & 15;

  v8f acc[NT];
  #pragma unroll
  for (int t = 0; t < NT; ++t) acc[t] = (v8f){};

  for (int kk = 0; kk < K; kk += 32) {
    // ---- A 16x32 f16 fragment: lane holds k = hsel*8 + {0..7, 16..23}
    const float* Arow = A + (size_t)(mt * 16 + rm) * K + kk + hsel * 8;
    float4 a0 = *(const float4*)(Arow);
    float4 a1 = *(const float4*)(Arow + 4);
    float4 a2 = *(const float4*)(Arow + 16);
    float4 a3 = *(const float4*)(Arow + 20);
    v16h a;
    a[0]  = (_Float16)a0.x; a[1]  = (_Float16)a0.y; a[2]  = (_Float16)a0.z; a[3]  = (_Float16)a0.w;
    a[4]  = (_Float16)a1.x; a[5]  = (_Float16)a1.y; a[6]  = (_Float16)a1.z; a[7]  = (_Float16)a1.w;
    a[8]  = (_Float16)a2.x; a[9]  = (_Float16)a2.y; a[10] = (_Float16)a2.z; a[11] = (_Float16)a2.w;
    a[12] = (_Float16)a3.x; a[13] = (_Float16)a3.y; a[14] = (_Float16)a3.z; a[15] = (_Float16)a3.w;

    #pragma unroll
    for (int t = 0; t < NT; ++t) {
      // ---- B 32x16 f16 fragment from LDS:
      // lane holds K = kk + hsel*8 + {0..7, 16..23}; N = lane&15
      const float* Bcol = Blds + (kk + hsel * 8) * Nc + (ng * NT + t) * 16 + rm;
      v16h b;
      #pragma unroll
      for (int j = 0; j < 16; ++j) b[j] = (_Float16)Bcol[(j + (j >= 8 ? 8 : 0)) * Nc];
      acc[t] = __builtin_amdgcn_wmma_f32_16x16x32_f16(false, a, false, b,
                                                      (short)0, acc[t], false, false);
    }
  }

  #pragma unroll
  for (int t = 0; t < NT; ++t) {
    int col = (ng * NT + t) * 16 + rm;
    float bv = HAS_BIAS ? bias[col] : 0.f;
    #pragma unroll
    for (int r = 0; r < 8; ++r) {
      // D layout: VGPR r -> row r (lanes 0-15) / row r+8 (lanes 16-31), N = lane&15
      int row = mt * 16 + r + hsel * 8;
      size_t o = (size_t)row * Nc + col;
      float v = acc[t][r] + bv;
      if (HAS_RES) v += res[o];
      C[o] = v;
    }
  }
}

// ------------------------------------------------------------ BatchNorm bits
__global__ void k_bn_stats(const float* __restrict__ x, float* __restrict__ s1,
                           float* __restrict__ s2, int M, int C) {
  int c = threadIdx.x;                 // blockDim.x == C
  float a = 0.f, b = 0.f;
  for (int r = blockIdx.x; r < M; r += gridDim.x) {
    float v = x[(size_t)r * C + c];
    a += v; b += v * v;
  }
  atomicAdd(&s1[c], a);
  atomicAdd(&s2[c], b);
}

__global__ void k_bn_apply(const float* __restrict__ x, float* __restrict__ y,
                           const float* __restrict__ s1, const float* __restrict__ s2,
                           const float* __restrict__ g, const float* __restrict__ b,
                           int M, int C, int leaky) {
  int i = blockIdx.x * blockDim.x + threadIdx.x;
  if (i >= M * C) return;
  int c = i % C;
  float invM = 1.f / (float)M;
  float mu  = s1[c] * invM;
  float var = s2[c] * invM - mu * mu;
  float v = (x[i] - mu) * rsqrtf(var + 1e-5f) * g[c] + b[c];
  y[i] = leaky ? (v > 0.f ? v : 0.01f * v) : fmaxf(v, 0.f);
}

// --------------------------------------------------------------- edge passes
// msg = relu(z[src] + edge_attr*We + be) + 1e-7  (always > 0)
__global__ void k_edge_max(const float* __restrict__ zsrc, const float* __restrict__ eattr,
                           const float* __restrict__ We, const float* __restrict__ be,
                           const int* __restrict__ src, const int* __restrict__ dst,
                           float* __restrict__ nmax) {
  int idx = blockIdx.x * blockDim.x + threadIdx.x;
  if (idx >= EE * HH) return;
  int e = idx >> 6, c = idx & 63;
  float ea = eattr[e] * We[c] + be[c];
  float m = fmaxf(zsrc[(size_t)src[e] * HH + c] + ea, 0.f) + 1e-7f;
  // positive floats: uint compare == float compare; buffer pre-zeroed
  atomicMax((unsigned int*)&nmax[(size_t)dst[e] * HH + c], __float_as_uint(m));
}

__global__ void k_edge_acc(const float* __restrict__ zsrc, const float* __restrict__ eattr,
                           const float* __restrict__ We, const float* __restrict__ be,
                           const int* __restrict__ src, const int* __restrict__ dst,
                           const float* __restrict__ tvec, int layer,
                           const float* __restrict__ nmax, float* __restrict__ nsum,
                           float* __restrict__ wsum) {
  int idx = blockIdx.x * blockDim.x + threadIdx.x;
  if (idx >= EE * HH) return;
  int e = idx >> 6, c = idx & 63;
  float t = tvec[layer];
  float ea = eattr[e] * We[c] + be[c];
  float m = fmaxf(zsrc[(size_t)src[e] * HH + c] + ea, 0.f) + 1e-7f;
  size_t o = (size_t)dst[e] * HH + c;
  float ex = __expf(t * (m - nmax[o]));
  atomicAdd(&nsum[o], ex);
  atomicAdd(&wsum[o], m * ex);
}

__global__ void k_combine(const float* __restrict__ wsum, const float* __restrict__ nsum,
                          const float* __restrict__ xd, float* __restrict__ out, int n) {
  int i = blockIdx.x * blockDim.x + threadIdx.x;
  if (i >= n) return;
  out[i] = wsum[i] / (nsum[i] + 1e-16f) + xd[i];
}

// ----------------------------------------------------------------- pooling
__global__ void k_count(const int* __restrict__ batch, float* __restrict__ cnt) {
  int n = blockIdx.x * blockDim.x + threadIdx.x;
  if (n < NN) atomicAdd(&cnt[batch[n]], 1.f);
}

__global__ void k_pool(const float* __restrict__ hf, const int* __restrict__ batch,
                       float* __restrict__ out) {
  int i = blockIdx.x * blockDim.x + threadIdx.x;
  if (i >= NN * HH) return;
  int n = i >> 6, c = i & 63;
  atomicAdd(&out[(size_t)batch[n] * HH + c], hf[i]);
}

__global__ void k_pooldiv(float* __restrict__ out, const float* __restrict__ cnt) {
  int i = blockIdx.x * blockDim.x + threadIdx.x;
  if (i >= GG * HH) return;
  out[i] /= fmaxf(cnt[i >> 6], 1.f);
}

// ------------------------------------------------------------------ launcher
extern "C" void kernel_launch(void* const* d_in, const int* in_sizes, int n_in,
                              void* d_out, int out_size, void* d_ws, size_t ws_size,
                              hipStream_t stream) {
  const float* x      = (const float*)d_in[0];
  const int*   ei     = (const int*)  d_in[1];
  const int*   batch  = (const int*)  d_in[2];
  const float* eattr  = (const float*)d_in[3];
  const float* W_src  = (const float*)d_in[4];
  const float* b_src  = (const float*)d_in[5];
  const float* W_dst  = (const float*)d_in[6];
  const float* b_dst  = (const float*)d_in[7];
  const float* W_edge = (const float*)d_in[8];
  const float* b_edge = (const float*)d_in[9];
  const float* tvec   = (const float*)d_in[10];
  const float* W1     = (const float*)d_in[11];
  const float* bias1  = (const float*)d_in[12];
  const float* bng    = (const float*)d_in[13];
  const float* bnb    = (const float*)d_in[14];
  const float* W2     = (const float*)d_in[15];
  const float* bias2  = (const float*)d_in[16];
  const float* ng     = (const float*)d_in[17];
  const float* nb     = (const float*)d_in[18];
  const int* src = ei;
  const int* dst = ei + EE;

  float* ws = (float*)d_ws;
  size_t o = 0;
  float* xs0  = ws + o; o += (size_t)NN * HH;
  float* xd0  = ws + o; o += (size_t)NN * HH;
  float* h    = ws + o; o += (size_t)NN * HH;
  float* z    = ws + o; o += (size_t)NN * HH;
  float* agg  = ws + o; o += (size_t)NN * HH;
  float* nmax = ws + o; o += (size_t)NN * HH;   // nmax/nsum/wsum contiguous
  float* nsum = ws + o; o += (size_t)NN * HH;
  float* wsum = ws + o; o += (size_t)NN * HH;
  float* hid  = ws + o; o += (size_t)NN * HH2;
  float* s1   = ws + o; o += 128;               // s1/s2 contiguous (256)
  float* s2   = ws + o; o += 128;
  float* cnt  = ws + o; o += GG;

  const int TB = 256;
  #define BLK(n) (((n) + TB - 1) / TB)
  const int EB = BLK(EE * HH);
  // wave = 16 x 64 output strip (NT=4): N=64 -> 1 group, N=128 -> 2 groups
  const int gw64  = ((NN / 16) * (HH  / 64) * 32 + TB - 1) / TB;
  const int gw128 = ((NN / 16) * (HH2 / 64) * 32 + TB - 1) / TB;
  const size_t lds_ab = (size_t)INC * HH  * 4;   // 128x64  B panel = 32 KB
  const size_t lds_w1 = (size_t)HH  * HH2 * 4;   // 64x128  B panel = 32 KB
  const size_t lds_w2 = (size_t)HH2 * HH  * 4;   // 128x64  B panel = 32 KB

  // ---- layer 0 input projections
  k_gemm_wmma<true, false, 4><<<gw64, TB, lds_ab, stream>>>(x, W_src, b_src, nullptr, xs0, NN, INC, HH);
  k_gemm_wmma<true, false, 4><<<gw64, TB, lds_ab, stream>>>(x, W_dst, b_dst, nullptr, xd0, NN, INC, HH);

  for (int i = 0; i < LL; ++i) {
    const float* zs;
    const float* zd;
    bool has_res = (i != 0);
    if (i == 0) { zs = xs0; zd = xd0; }
    else {
      // z = leaky_relu(bn(h; norm_g[i], norm_b[i]))
      k_zero<<<1, 256, 0, stream>>>(s1, 256);
      k_bn_stats<<<128, HH, 0, stream>>>(h, s1, s2, NN, HH);
      k_bn_apply<<<BLK(NN * HH), TB, 0, stream>>>(h, z, s1, s2,
                                                  ng + i * HH, nb + i * HH, NN, HH, 1);
      zs = z; zd = z;
    }
    // segment-softmax aggregation (2 edge passes)
    k_zero<<<BLK(3 * NN * HH), TB, 0, stream>>>(nmax, 3 * NN * HH);
    k_edge_max<<<EB, TB, 0, stream>>>(zs, eattr, W_edge + i * HH, b_edge + i * HH,
                                      src, dst, nmax);
    k_edge_acc<<<EB, TB, 0, stream>>>(zs, eattr, W_edge + i * HH, b_edge + i * HH,
                                      src, dst, tvec, i, nmax, nsum, wsum);
    k_combine<<<BLK(NN * HH), TB, 0, stream>>>(wsum, nsum, zd, agg, NN * HH);

    // MLP: Linear -> BN -> ReLU -> Linear (+ residual for i>=1)
    k_gemm_wmma<true, false, 4><<<gw128, TB, lds_w1, stream>>>(agg, W1 + (size_t)i * HH * HH2,
                                                               bias1 + i * HH2, nullptr, hid,
                                                               NN, HH, HH2);
    k_zero<<<1, 256, 0, stream>>>(s1, 256);
    k_bn_stats<<<128, HH2, 0, stream>>>(hid, s1, s2, NN, HH2);
    k_bn_apply<<<BLK(NN * HH2), TB, 0, stream>>>(hid, hid, s1, s2,
                                                 bng + i * HH2, bnb + i * HH2, NN, HH2, 0);
    if (has_res)
      k_gemm_wmma<true, true, 4><<<gw64, TB, lds_w2, stream>>>(hid, W2 + (size_t)i * HH2 * HH,
                                                               bias2 + i * HH, h, h, NN, HH2, HH);
    else
      k_gemm_wmma<true, false, 4><<<gw64, TB, lds_w2, stream>>>(hid, W2 + (size_t)i * HH2 * HH,
                                                                bias2 + i * HH, nullptr, h,
                                                                NN, HH2, HH);
  }

  // ---- final norm (layer-0 params) + leaky_relu + mean pool per graph
  k_zero<<<1, 256, 0, stream>>>(s1, 256);
  k_bn_stats<<<128, HH, 0, stream>>>(h, s1, s2, NN, HH);
  k_bn_apply<<<BLK(NN * HH), TB, 0, stream>>>(h, z, s1, s2, ng, nb, NN, HH, 1);

  k_zero<<<BLK(GG * HH), TB, 0, stream>>>((float*)d_out, GG * HH);
  k_zero<<<1, GG, 0, stream>>>(cnt, GG);
  k_count<<<BLK(NN), TB, 0, stream>>>(batch, cnt);
  k_pool<<<BLK(NN * HH), TB, 0, stream>>>(z, batch, (float*)d_out);
  k_pooldiv<<<BLK(GG * HH), TB, 0, stream>>>((float*)d_out, cnt);
  #undef BLK
}
